// TransformerSymile_79963701117618
// MI455X (gfx1250) — matrix operand order repaired and checked by reading, not compiled, not run on hardware
//
#include <hip/hip_runtime.h>
#include <math.h>

// ---------------- problem constants ----------------
static constexpr int kB    = 32;
static constexpr int kSE   = 128;
static constexpr int kS    = 384;          // 3*128
static constexpr int kD    = 512;
static constexpr int kH    = 8;
static constexpr int kL    = 4;
static constexpr int kDFF  = 512;
static constexpr int kPOUT = 128;
static constexpr int kF    = kD * kS;      // 196608
static constexpr int kBS   = kB * kS;      // 12288
static constexpr int kKSPLIT = 32;         // split-K chunks for proj GEMM
static constexpr int kKCHUNK = kF / kKSPLIT; // 6144

typedef __attribute__((ext_vector_type(16))) __bf16 v16bf;
typedef __attribute__((ext_vector_type(8)))  float  v8f;

struct alignas(16) U4 { unsigned int x, y, z, w; };
union BFrag { unsigned int u[8]; U4 q[2]; v16bf v; };

// native f32 -> bf16 (RNE) via hardware cvt, not manual bit-twiddling
__device__ __forceinline__ unsigned short f2bf(float f) {
  __bf16 h = (__bf16)f;
  unsigned short u;
  __builtin_memcpy(&u, &h, 2);
  return u;
}
__device__ __forceinline__ unsigned int pack2bf(float a, float b) {
  return (unsigned int)f2bf(a) | ((unsigned int)f2bf(b) << 16);
}
__device__ __forceinline__ float geluf(float x) {
  return 0.5f * x * (1.0f + erff(x * 0.70710678118654752f));
}
// fast sigmoid: v_exp + v_rcp (1 ulp), plenty for a bf16 WMMA operand
__device__ __forceinline__ float sigmoidf_fast(float x) {
  return __builtin_amdgcn_rcpf(1.0f + __expf(-x));
}

// ---------------- concat embeddings ----------------
__global__ __launch_bounds__(256) void concat_k(const float* __restrict__ e0,
                                                const float* __restrict__ e1,
                                                const float* __restrict__ e2,
                                                float* __restrict__ x) {
  int i = blockIdx.x * 256 + threadIdx.x;               // over kB*kS*kD
  int b = i / (kS * kD);
  int rem = i - b * (kS * kD);
  int s = rem / kD;
  int d = rem - s * kD;
  float v;
  if (s < kSE)            v = e0[(b * kSE + s) * kD + d];
  else if (s < 2 * kSE)   v = e1[(b * kSE + (s - kSE)) * kD + d];
  else                    v = e2[(b * kSE + (s - 2 * kSE)) * kD + d];
  x[i] = v;
}

// ---------------- weight convert: fp32 [N,K] -> pair-packed bf16 [K/2][N] ----------------
__global__ __launch_bounds__(256) void wconv_k(const float* __restrict__ W,
                                               unsigned int* __restrict__ Bp,
                                               int N, int K, int total) {
  int i = blockIdx.x * 256 + threadIdx.x;               // over (K/2)*N
  if (i >= total) return;
  int pk = i / N;
  int n  = i - pk * N;
  size_t base = (size_t)n * K + 2 * pk;
  Bp[i] = pack2bf(W[base], W[base + 1]);
}

// ---------------- LayerNorm over D=512 rows, bf16 out ----------------
__global__ __launch_bounds__(256) void ln_rows_k(const float* __restrict__ x,
                                                 const float* __restrict__ w,
                                                 const float* __restrict__ b,
                                                 unsigned short* __restrict__ out) {
  __shared__ float red[256];
  int row = blockIdx.x, tid = threadIdx.x;
  size_t base = (size_t)row * kD;
  float v0 = x[base + tid];
  float v1 = x[base + 256 + tid];
  red[tid] = v0 + v1; __syncthreads();
  for (int o = 128; o > 0; o >>= 1) { if (tid < o) red[tid] += red[tid + o]; __syncthreads(); }
  float mean = red[0] * (1.0f / 512.0f);
  __syncthreads();
  float d0 = v0 - mean, d1 = v1 - mean;
  red[tid] = d0 * d0 + d1 * d1; __syncthreads();
  for (int o = 128; o > 0; o >>= 1) { if (tid < o) red[tid] += red[tid + o]; __syncthreads(); }
  float rs = rsqrtf(red[0] * (1.0f / 512.0f) + 1e-5f);
  out[base + tid]       = f2bf(d0 * rs * w[tid] + b[tid]);
  out[base + 256 + tid] = f2bf(d1 * rs * w[256 + tid] + b[256 + tid]);
}

// ---- fragment loaders: two 16B loads each -> contiguous VGPR quads for WMMA ----
// A 16x32 bf16 frag: k-pairs (0..3)+4h and (8..11)+4h are contiguous -> two b128 loads.
__device__ __forceinline__ void loadA(BFrag& a, const unsigned int* rowBase, int half) {
  a.q[0] = *(const U4*)(rowBase + 4 * half);
  a.q[1] = *(const U4*)(rowBase + 8 + 4 * half);
}
// B 32x16 bf16 frag from column-major storage: k-pairs (0..3)+4h and (8..11)+4h, matching loadA.
__device__ __forceinline__ void loadB(BFrag& b, const unsigned int* colBase, int half) {
  b.q[0] = *(const U4*)(colBase + 4 * half);
  b.q[1] = *(const U4*)(colBase + 8 + 4 * half);
}

// ---------------- generic WMMA GEMM: C[M,N] = A[M,K](bf16) x Bp[K/2,N](packed bf16) ----------------
// 256 threads = 8 waves; block tile 64x64; wave tile 16x32; K stepped by 32.
template <int BIAS, int ACT_GELU, int RESID, int OUTBF>
__global__ __launch_bounds__(256) void gemm_k(const unsigned short* __restrict__ A,
                                              const unsigned int* __restrict__ Bp,
                                              const float* __restrict__ bias,
                                              const float* __restrict__ resid,
                                              float* __restrict__ outF,
                                              unsigned short* __restrict__ outB,
                                              int M, int N, int K) {
  __shared__ unsigned int sA[64][20];    // [row][k-pair], stride 80B (16B-aligned quads)
  __shared__ unsigned int sBt[64][20];   // [col][k-pair] (transposed for contiguous frags)
  int m0 = blockIdx.x * 64, n0 = blockIdx.y * 64;
  int tid = threadIdx.x;
  int w = tid >> 5, l = tid & 31, half = l >> 4, ln = l & 15;
  int wm = (w & 3) * 16, wn = (w >> 2) * 32;
  const unsigned int* A32 = (const unsigned int*)A;
  v8f c0 = {}, c1 = {};
  for (int k0 = 0; k0 < K; k0 += 32) {
    #pragma unroll
    for (int i = 0; i < 4; ++i) {
      int e = tid + i * 256; int r = e >> 4, p = e & 15;
      sA[r][p] = A32[((size_t)(m0 + r) * K + k0) / 2 + p];
    }
    #pragma unroll
    for (int i = 0; i < 4; ++i) {
      int e = tid + i * 256; int p = e >> 6, n = e & 63;   // coalesced global read
      sBt[n][p] = Bp[(size_t)(k0 / 2 + p) * N + n0 + n];
    }
    __syncthreads();
    BFrag a, b0, b1;
    loadA(a, &sA[wm + ln][0], half);
    loadB(b0, &sBt[wn + ln][0], half);
    loadB(b1, &sBt[wn + 16 + ln][0], half);
    c0 = __builtin_amdgcn_wmma_f32_16x16x32_bf16(false, a.v, false, b0.v, (short)0, c0, false, false);
    c1 = __builtin_amdgcn_wmma_f32_16x16x32_bf16(false, a.v, false, b1.v, (short)0, c1, false, false);
    __syncthreads();
  }
  #pragma unroll
  for (int r = 0; r < 8; ++r) {
    int row = m0 + wm + r + 8 * half;
    #pragma unroll
    for (int nt = 0; nt < 2; ++nt) {
      int col = n0 + wn + nt * 16 + ln;
      float v = (nt ? c1[r] : c0[r]);
      if (BIAS)     v += bias[col];
      if (ACT_GELU) v = geluf(v);
      size_t idx = (size_t)row * N + col;
      if (RESID)    v += resid[idx];
      if (OUTBF)    outB[idx] = f2bf(v);
      else          outF[idx] = v;
    }
  }
}

// ---------------- fused sigmoid attention per (b,h) ----------------
// grid = B*H blocks, 256 threads (8 waves, 3 q-tiles each). Dynamic LDS:
//   sK  [384][36] u32 (dh pairs, padded stride)
//   sVt [64][200] u32 ([dh][t-pair], transposed+padded for contiguous b128 frags)
//   sS  8 x (16x32) bf16 wave-private score scratch
static constexpr int kKStr = 36;
static constexpr int kVStr = 200;
static constexpr int kAttnLdsU32 = kS * kKStr + 64 * kVStr + 2048; // 28672 u32
__global__ __launch_bounds__(256) void attn_k(const unsigned short* __restrict__ qkv,
                                              unsigned short* __restrict__ o) {
  extern __shared__ unsigned int smem[];
  unsigned int* sK  = smem;                       // kS*kKStr
  unsigned int* sVt = smem + kS * kKStr;          // 64*kVStr
  unsigned short* sS = (unsigned short*)(smem + kS * kKStr + 64 * kVStr); // 4096 u16

  int bh = blockIdx.x;
  int b = bh >> 3, h = bh & 7;
  int tid = threadIdx.x;
  int w = tid >> 5, l = tid & 31, half = l >> 4, ln = l & 15;
  const unsigned int* q32 = (const unsigned int*)qkv;
  size_t baseE = (size_t)b * kS * 1536;           // element base of this batch row

  for (int i = tid; i < kS * 32; i += 256) {      // K: [t][dh-pair]
    int t = i >> 5, pd = i & 31;
    sK[t * kKStr + pd] = q32[(baseE + (size_t)t * 1536 + 512 + h * 64) / 2 + pd];
  }
  for (int i = tid; i < 64 * 192; i += 256) {     // V: [dh][t-pair]
    int dh = i / 192, pt = i - dh * 192;
    size_t e0 = baseE + (size_t)(2 * pt) * 1536 + 1024 + h * 64 + dh;
    sVt[dh * kVStr + pt] = (unsigned int)qkv[e0] | ((unsigned int)qkv[e0 + 1536] << 16);
  }
  __syncthreads();

  const float scale = 0.125f;                     // 1/sqrt(64)
  unsigned short* myS = sS + w * 512;             // wave-private 16x32 bf16
  const unsigned int* myS32 = (const unsigned int*)myS;

  for (int qi = 0; qi < 3; ++qi) {
    int qt = w + 8 * qi;                          // q tile index (0..23)
    int srow = qt * 16 + ln;
    const unsigned int* qrow = q32 + (baseE + (size_t)srow * 1536 + h * 64) / 2;
    BFrag aq0, aq1;
    loadA(aq0, qrow, half);                       // dh 0..31  (global b128 x2)
    loadA(aq1, qrow + 16, half);                  // dh 32..63
    v8f o0 = {}, o1 = {}, o2 = {}, o3 = {};
    for (int t0 = 0; t0 < kS; t0 += 32) {
      // two 16x16 sigmoid(score) tiles -> wave-private LDS (bf16, t-major pairs)
      #pragma unroll
      for (int tt = 0; tt < 2; ++tt) {
        v8f sc = {};
        BFrag bk;
        const unsigned int* kt = sK + (t0 + tt * 16 + ln) * kKStr;
        loadB(bk, kt, half);                      // dh block 0
        sc = __builtin_amdgcn_wmma_f32_16x16x32_bf16(false, aq0.v, false, bk.v, (short)0, sc, false, false);
        loadB(bk, kt + 16, half);                 // dh block 1
        sc = __builtin_amdgcn_wmma_f32_16x16x32_bf16(false, aq1.v, false, bk.v, (short)0, sc, false, false);
        #pragma unroll
        for (int r = 0; r < 8; ++r) {
          float p = sigmoidf_fast(sc[r] * scale);
          myS[(r + 8 * half) * 32 + tt * 16 + ln] = f2bf(p);
        }
      }
      // reload as A-frag (16 rows x 32 t); same-wave LDS ordering guarantees visibility
      BFrag as;
      loadA(as, myS32 + ln * 16, half);
      int pt0 = t0 >> 1;
      BFrag bv;
      loadB(bv, sVt + (size_t)ln * kVStr + pt0, half);
      o0 = __builtin_amdgcn_wmma_f32_16x16x32_bf16(false, as.v, false, bv.v, (short)0, o0, false, false);
      loadB(bv, sVt + (size_t)(16 + ln) * kVStr + pt0, half);
      o1 = __builtin_amdgcn_wmma_f32_16x16x32_bf16(false, as.v, false, bv.v, (short)0, o1, false, false);
      loadB(bv, sVt + (size_t)(32 + ln) * kVStr + pt0, half);
      o2 = __builtin_amdgcn_wmma_f32_16x16x32_bf16(false, as.v, false, bv.v, (short)0, o2, false, false);
      loadB(bv, sVt + (size_t)(48 + ln) * kVStr + pt0, half);
      o3 = __builtin_amdgcn_wmma_f32_16x16x32_bf16(false, as.v, false, bv.v, (short)0, o3, false, false);
    }
    #pragma unroll
    for (int r = 0; r < 8; ++r) {
      int row = qt * 16 + r + 8 * half;
      size_t ob = (size_t)(b * kS + row) * kD + h * 64 + ln;
      o[ob]      = f2bf(o0[r]);
      o[ob + 16] = f2bf(o1[r]);
      o[ob + 32] = f2bf(o2[r]);
      o[ob + 48] = f2bf(o3[r]);
    }
  }
}

// ---------------- proj-head LayerNorm over F=196608 (32 rows) ----------------
__global__ __launch_bounds__(1024) void projln_k(const float* __restrict__ x,
                                                 const float* __restrict__ w,
                                                 const float* __restrict__ b,
                                                 unsigned short* __restrict__ out) {
  __shared__ float red[1024];
  int row = blockIdx.x, tid = threadIdx.x;
  size_t base = (size_t)row * kF;
  float s = 0.f, ss = 0.f;
  for (int i = tid; i < kF; i += 1024) { float v = x[base + i]; s += v; ss += v * v; }
  red[tid] = s; __syncthreads();
  for (int o = 512; o > 0; o >>= 1) { if (tid < o) red[tid] += red[tid + o]; __syncthreads(); }
  float mean = red[0] * (1.0f / (float)kF);
  __syncthreads();
  red[tid] = ss; __syncthreads();
  for (int o = 512; o > 0; o >>= 1) { if (tid < o) red[tid] += red[tid + o]; __syncthreads(); }
  float var = red[0] * (1.0f / (float)kF) - mean * mean;
  float rs = rsqrtf(var + 1e-5f);
  for (int i = tid; i < kF; i += 1024)
    out[base + i] = f2bf((x[base + i] - mean) * rs * w[i] + b[i]);
}

// ---------------- proj GEMM: part[kc][32][512] = zln[32,F] x ph_W1^T chunk (on-the-fly bf16) ----------------
// grid (N/64=8, KSPLIT=32); streams the 402MB ph_W1 exactly once -> HBM bound (~17us @ 23.3TB/s).
__global__ __launch_bounds__(256) void proj_gemm_k(const unsigned short* __restrict__ zln,
                                                   const float* __restrict__ phW1,
                                                   float* __restrict__ part) {
  __shared__ unsigned int sA[32][20];
  __shared__ unsigned int sBt[64][20];
  int n0 = blockIdx.x * 64;
  int kc = blockIdx.y;
  int tid = threadIdx.x;
  int w = tid >> 5, l = tid & 31, half = l >> 4, ln = l & 15;
  int mt = w & 1, nt = w >> 1;                 // 2 x 4 wave grid, one 16x16 tile each
  const unsigned int* z32 = (const unsigned int*)zln;
  v8f c = {};
  int kend = kc * kKCHUNK + kKCHUNK;
  for (int k0 = kc * kKCHUNK; k0 < kend; k0 += 32) {
    #pragma unroll
    for (int i = 0; i < 2; ++i) {
      int e = tid + i * 256; int r = e >> 4, p = e & 15;
      sA[r][p] = z32[((size_t)r * kF + k0) / 2 + p];
    }
    #pragma unroll
    for (int i = 0; i < 4; ++i) {
      int e = tid + i * 256; int p = e >> 6, n = e & 63;
      size_t gb = (size_t)(n0 + n) * kF + k0 + 2 * p;
      sBt[n][p] = pack2bf(phW1[gb], phW1[gb + 1]);
    }
    __syncthreads();
    BFrag a, bb;
    loadA(a, &sA[mt * 16 + ln][0], half);
    loadB(bb, &sBt[nt * 16 + ln][0], half);
    c = __builtin_amdgcn_wmma_f32_16x16x32_bf16(false, a.v, false, bb.v, (short)0, c, false, false);
    __syncthreads();
  }
  #pragma unroll
  for (int r = 0; r < 8; ++r) {
    int row = mt * 16 + r + 8 * half;
    int col = n0 + nt * 16 + ln;
    part[((size_t)kc * 32 + row) * 512 + col] = c[r];
  }
}

// ---------------- deterministic split-K reduce + bias + GELU ----------------
__global__ __launch_bounds__(256) void reduce_gelu_k(const float* __restrict__ part,
                                                     const float* __restrict__ b1,
                                                     float* __restrict__ p1) {
  int i = blockIdx.x * 256 + threadIdx.x;      // over 32*512
  float s = b1[i & 511];
  for (int kc = 0; kc < kKSPLIT; ++kc) s += part[(size_t)kc * (32 * 512) + i];
  p1[i] = geluf(s);
}

// ---------------- final tiny GEMM [32,512] x ph_W2^T -> [32,128] (fp32) ----------------
__global__ __launch_bounds__(256) void gemm2_k(const float* __restrict__ p1,
                                               const float* __restrict__ W2,
                                               const float* __restrict__ b2,
                                               float* __restrict__ out) {
  int i = blockIdx.x * 256 + threadIdx.x;      // over 32*128
  int b = i >> 7, p = i & 127;
  float s = b2[p];
  const float* a = p1 + b * 512;
  const float* w = W2 + p * 512;
  for (int k = 0; k < 512; ++k) s += a[k] * w[k];
  out[i] = s;
}

// ---------------- host orchestration ----------------
extern "C" void kernel_launch(void* const* d_in, const int* in_sizes, int n_in,
                              void* d_out, int out_size, void* d_ws, size_t ws_size,
                              hipStream_t stream) {
  (void)in_sizes; (void)n_in; (void)out_size; (void)ws_size;
  const float* emb0   = (const float*)d_in[0];
  const float* emb1   = (const float*)d_in[1];
  const float* emb2   = (const float*)d_in[2];
  const float* W_in   = (const float*)d_in[3];
  const float* b_in   = (const float*)d_in[4];
  const float* W_out  = (const float*)d_in[5];
  const float* b_out  = (const float*)d_in[6];
  const float* ln1_w  = (const float*)d_in[7];
  const float* ln1_b  = (const float*)d_in[8];
  const float* ln2_w  = (const float*)d_in[9];
  const float* ln2_b  = (const float*)d_in[10];
  const float* W1     = (const float*)d_in[11];
  const float* b1     = (const float*)d_in[12];
  const float* W2     = (const float*)d_in[13];
  const float* b2     = (const float*)d_in[14];
  const float* phlnw  = (const float*)d_in[15];
  const float* phlnb  = (const float*)d_in[16];
  const float* phW1   = (const float*)d_in[17];
  const float* phb1   = (const float*)d_in[18];
  const float* phW2   = (const float*)d_in[19];
  const float* phb2   = (const float*)d_in[20];

  // workspace carve (all 256B aligned by construction)
  char* ws = (char*)d_ws;
  float*          x     = (float*)ws;                       ws += (size_t)kBS * kD * 4;       // 25.2 MB
  unsigned short* hbuf  = (unsigned short*)ws;              ws += (size_t)kBS * kD * 2;       // 12.6 MB (LN out / zln)
  unsigned short* qkv   = (unsigned short*)ws;              ws += (size_t)kBS * 3 * kD * 2;   // 37.7 MB
  unsigned short* tbuf  = (unsigned short*)ws;              ws += (size_t)kBS * kD * 2;       // 12.6 MB (attn o / ffn1)
  unsigned int*   WinT  = (unsigned int*)ws;                ws += (size_t)kL * (kD / 2) * (3 * kD) * 4;
  unsigned int*   WoutT = (unsigned int*)ws;                ws += (size_t)kL * (kD / 2) * kD * 4;
  unsigned int*   W1T   = (unsigned int*)ws;                ws += (size_t)kL * (kD / 2) * kDFF * 4;
  unsigned int*   W2T   = (unsigned int*)ws;                ws += (size_t)kL * (kDFF / 2) * kD * 4;
  float*          part  = (float*)ws;                       ws += (size_t)kKSPLIT * 32 * 512 * 4;
  float*          p1    = (float*)ws;                       ws += (size_t)32 * 512 * 4;

  // convert/transpose/pair-pack weights to bf16 [K/2][N]
  for (int l = 0; l < kL; ++l) {
    { int tot = (kD / 2) * (3 * kD);
      wconv_k<<<(tot + 255) / 256, 256, 0, stream>>>(W_in + (size_t)l * 3 * kD * kD,
                                                     WinT + (size_t)l * tot, 3 * kD, kD, tot); }
    { int tot = (kD / 2) * kD;
      wconv_k<<<(tot + 255) / 256, 256, 0, stream>>>(W_out + (size_t)l * kD * kD,
                                                     WoutT + (size_t)l * tot, kD, kD, tot); }
    { int tot = (kD / 2) * kDFF;
      wconv_k<<<(tot + 255) / 256, 256, 0, stream>>>(W1 + (size_t)l * kDFF * kD,
                                                     W1T + (size_t)l * tot, kDFF, kD, tot); }
    { int tot = (kDFF / 2) * kD;
      wconv_k<<<(tot + 255) / 256, 256, 0, stream>>>(W2 + (size_t)l * kD * kDFF,
                                                     W2T + (size_t)l * tot, kD, kDFF, tot); }
  }

  concat_k<<<(kBS * kD) / 256, 256, 0, stream>>>(emb0, emb1, emb2, x);

  const unsigned int attnLds = kAttnLdsU32 * 4;  // 114,688 B dynamic LDS (<< 320KB/WGP)
  for (int l = 0; l < kL; ++l) {
    ln_rows_k<<<kBS, 256, 0, stream>>>(x, ln1_w + l * kD, ln1_b + l * kD, hbuf);
    gemm_k<1, 0, 0, 1><<<dim3(kBS / 64, (3 * kD) / 64), 256, 0, stream>>>(
        hbuf, WinT + (size_t)l * (kD / 2) * (3 * kD), b_in + l * 3 * kD,
        nullptr, nullptr, qkv, kBS, 3 * kD, kD);
    attn_k<<<kB * kH, 256, attnLds, stream>>>(qkv, tbuf);
    gemm_k<1, 0, 1, 0><<<dim3(kBS / 64, kD / 64), 256, 0, stream>>>(
        tbuf, WoutT + (size_t)l * (kD / 2) * kD, b_out + l * kD,
        x, x, nullptr, kBS, kD, kD);
    ln_rows_k<<<kBS, 256, 0, stream>>>(x, ln2_w + l * kD, ln2_b + l * kD, hbuf);
    gemm_k<1, 1, 0, 1><<<dim3(kBS / 64, kDFF / 64), 256, 0, stream>>>(
        hbuf, W1T + (size_t)l * (kD / 2) * kDFF, b1 + l * kDFF,
        nullptr, nullptr, tbuf, kBS, kDFF, kD);
    gemm_k<1, 0, 1, 0><<<dim3(kBS / 64, kD / 64), 256, 0, stream>>>(
        tbuf, W2T + (size_t)l * (kDFF / 2) * kD, b2 + l * kD,
        x, x, nullptr, kBS, kD, kDFF);
  }

  // projection head
  projln_k<<<kB, 1024, 0, stream>>>(x, phlnw, phlnb, hbuf);
  proj_gemm_k<<<dim3(kD / 64, kKSPLIT), 256, 0, stream>>>(hbuf, phW1, part);
  reduce_gelu_k<<<(32 * 512) / 256, 256, 0, stream>>>(part, phb1, p1);
  gemm2_k<<<(kB * kPOUT) / 256, 256, 0, stream>>>(p1, phW2, phb2, (float*)d_out);
}
